// LLaDA2MoeSparseMoeBlock_9191230013600
// MI455X (gfx1250) — compile-verified, matrix operand order checked by reading, not run on hardware
//
#include <hip/hip_runtime.h>
#include <hip/hip_bf16.h>
#include <math.h>

// ---------------------------------------------------------------------------
// LLaDA2 MoE block for MI455X (gfx1250, wave32, WMMA, async LDS staging).
// - all GEMMs: v_wmma_f32_16x16x32_bf16, f32 accumulate
// - weights converted once to bf16 AND transposed to [N][K] so every WMMA
//   operand is K-contiguous per lane -> staging/frag traffic is pure b128
// - double-buffered LDS tiles staged with global_load_async_to_lds_b128
//   (ASYNCcnt) when available, sync b128 copies otherwise
// ---------------------------------------------------------------------------

typedef __bf16 bf16;
typedef __attribute__((ext_vector_type(4)))  float  v4f;
typedef __attribute__((ext_vector_type(8)))  float  v8f;
typedef __attribute__((ext_vector_type(4)))  __bf16 v4bf;
typedef __attribute__((ext_vector_type(8)))  __bf16 v8bf;
typedef __attribute__((ext_vector_type(16))) __bf16 v16bf;
typedef __attribute__((ext_vector_type(4)))  int    v4i;

#define TOKENS 4096
#define HDIM   1024
#define IDIM   512
#define NEXP   32
#define TOPK   4
#define ROUTED_SCALE 2.5f

#define BM 64
#define BN 128
#define BK 32
#define NTHREADS 256
#define LDSPAD 48          // LDS row stride (elems) = 96B; keeps 16B alignment
#define CAP 18432          // 16384 + 32*63 padded expert-sorted rows
#define MAX_TILES 288      // sum ceil(c_e/64) <= 16384/64 + 32

#if defined(__has_builtin)
#  if __has_builtin(__builtin_amdgcn_global_load_async_to_lds_b128)
#    define HAVE_ASYNC 1
#  endif
#  if __has_builtin(__builtin_amdgcn_s_wait_asynccnt)
#    define HAVE_WAIT_ASYNC_BUILTIN 1
#  endif
#endif
#ifndef HAVE_ASYNC
#  define HAVE_ASYNC 0
#endif

// typed address-space pointers for the async-copy builtin:
// param0 = int4 addrspace(1)* (global), param1 = int4 addrspace(3)* (LDS)
typedef __attribute__((address_space(1))) v4i g_v4i;
typedef __attribute__((address_space(3))) v4i l_v4i;
#define AS1P(p) ((g_v4i*)(p))
#define AS3P(p) ((l_v4i*)(p))

__device__ __forceinline__ void wait_async_lds() {
#if HAVE_ASYNC
#  if defined(HAVE_WAIT_ASYNC_BUILTIN)
  __builtin_amdgcn_s_wait_asynccnt(0);
#  else
  asm volatile("s_wait_asynccnt 0x0" ::: "memory");
#  endif
#endif
}

__device__ __forceinline__ v8f zero8f() {
  v8f r;
#pragma unroll
  for (int i = 0; i < 8; ++i) r[i] = 0.f;
  return r;
}
__device__ __forceinline__ v8bf zero8bf() {
  v8bf r;
#pragma unroll
  for (int i = 0; i < 8; ++i) r[i] = (__bf16)0.f;
  return r;
}

// 16-elem bf16 fragment per ISA 16-bit A/B layout: lane half selects
// K{kb..kb+7} and K{kb+16..kb+23}; two ds_load_b128 from one LDS row.
__device__ __forceinline__ v16bf load_frag(const bf16* row, int kb) {
  v8bf lo = *(const v8bf*)(row + kb);
  v8bf hi = *(const v8bf*)(row + kb + 16);
  return __builtin_shufflevector(lo, hi, 0, 1, 2, 3, 4, 5, 6, 7,
                                 8, 9, 10, 11, 12, 13, 14, 15);
}

// ---------------------------------------------------------------------------
// fp32 -> bf16 straight conversion (activations)
// ---------------------------------------------------------------------------
__global__ __launch_bounds__(256) void cvt_bf16_kernel(const float* __restrict__ src,
                                                       bf16* __restrict__ dst, int n4) {
  int i = blockIdx.x * blockDim.x + threadIdx.x;
  if (i >= n4) return;
  v4f v = *(const v4f*)(src + (size_t)i * 4);
  v4bf o;
  o.x = (__bf16)v.x; o.y = (__bf16)v.y; o.z = (__bf16)v.z; o.w = (__bf16)v.w;
  *(v4bf*)(dst + (size_t)i * 4) = o;
}

// fp32 [mat][K][N] -> bf16 [mat][N][K] transpose-convert (weights, once/call)
__global__ __launch_bounds__(256) void cvt_tr_kernel(const float* __restrict__ src,
                                                     bf16* __restrict__ dst,
                                                     int K, int N, int total8) {
  int tid = blockIdx.x * blockDim.x + threadIdx.x;
  if (tid >= total8) return;
  int kc = K >> 3;
  int mat = tid / (N * kc);
  int rem = tid - mat * (N * kc);
  int n = rem / kc;
  int k = (rem - n * kc) * 8;
  const float* s = src + (size_t)mat * K * N + (size_t)k * N + n;
  v8bf o;
#pragma unroll
  for (int j = 0; j < 8; ++j) o[j] = (__bf16)s[(size_t)j * N];
  *(v8bf*)(dst + (size_t)mat * K * N + (size_t)n * K + k) = o;
}

// ---------------------------------------------------------------------------
// Router: 1 wave per token, lane = expert (E == 32 == wave32)
// ---------------------------------------------------------------------------
__global__ __launch_bounds__(256) void gate_kernel(
    const float* __restrict__ x, const float* __restrict__ gw,
    const float* __restrict__ bias, float* __restrict__ logits_out,
    int* __restrict__ topk_out, float* __restrict__ tw_out,
    int* __restrict__ counts) {
  int wave = threadIdx.x >> 5;
  int lane = threadIdx.x & 31;
  int t = blockIdx.x * 8 + wave;
  if (t >= TOKENS) return;

  const float* xr = x + (size_t)t * HDIM;
  const float* gr = gw + (size_t)lane * HDIM;
  float acc = 0.f;
  for (int k = 0; k < HDIM; k += 4) {
    v4f xv = *(const v4f*)(xr + k);
    v4f gv = *(const v4f*)(gr + k);
    acc += xv.x * gv.x + xv.y * gv.y + xv.z * gv.z + xv.w * gv.w;
  }
  float logit = acc;
  logits_out[(size_t)t * NEXP + lane] = logit;

  float score = 1.f / (1.f + __expf(-logit));
  float srt = score + bias[lane];

  // group (8 experts) top-2 sum
  float m1v = srt; int m1e = lane;
#pragma unroll
  for (int off = 1; off < 8; off <<= 1) {
    float ov = __shfl_xor(m1v, off); int oe = __shfl_xor(m1e, off);
    if (ov > m1v || (ov == m1v && oe < m1e)) { m1v = ov; m1e = oe; }
  }
  float m2v = (lane == m1e) ? -__builtin_inff() : srt;
#pragma unroll
  for (int off = 1; off < 8; off <<= 1) m2v = fmaxf(m2v, __shfl_xor(m2v, off));
  float gscore = m1v + m2v;

  float gsv[4];
  gsv[0] = __shfl(gscore, 0);  gsv[1] = __shfl(gscore, 8);
  gsv[2] = __shfl(gscore, 16); gsv[3] = __shfl(gscore, 24);
  int g1 = 0; float b1 = gsv[0];
#pragma unroll
  for (int g = 1; g < 4; ++g) if (gsv[g] > b1) { b1 = gsv[g]; g1 = g; }
  int g2 = -1; float b2 = -__builtin_inff();
#pragma unroll
  for (int g = 0; g < 4; ++g) if (g != g1 && gsv[g] > b2) { b2 = gsv[g]; g2 = g; }
  int myg = lane >> 3;
  float masked = (myg == g1 || myg == g2) ? srt : -__builtin_inff();

  int   idx[4];
  float sc[4];
#pragma unroll
  for (int j = 0; j < 4; ++j) {
    float mv = masked; int me = lane;
#pragma unroll
    for (int off = 1; off < 32; off <<= 1) {
      float ov = __shfl_xor(mv, off); int oe = __shfl_xor(me, off);
      if (ov > mv || (ov == mv && oe < me)) { mv = ov; me = oe; }
    }
    idx[j] = me;
    sc[j] = __shfl(score, me);
    if (lane == me) masked = -__builtin_inff();
  }
  float ssum = sc[0] + sc[1] + sc[2] + sc[3] + 1e-20f;
  if (lane == 0) {
#pragma unroll
    for (int j = 0; j < 4; ++j) {
      topk_out[(size_t)t * TOPK + j] = idx[j];
      tw_out[(size_t)t * TOPK + j] = sc[j] / ssum * ROUTED_SCALE;
      atomicAdd(&counts[idx[j]], 1);
    }
  }
}

// ---------------------------------------------------------------------------
// Device-side scheduler: padded per-expert offsets + M-tile work list
// ---------------------------------------------------------------------------
__global__ void builder_kernel(const int* __restrict__ counts, int* __restrict__ offsets,
                               int* __restrict__ fill, int2* __restrict__ schedule,
                               int* __restrict__ ntiles) {
  int e = threadIdx.x;
  if (e < NEXP) fill[e] = 0;
  if (threadIdx.x == 0) {
    int off = 0, nt = 0;
    for (int i = 0; i < NEXP; ++i) {
      int c = counts[i];
      offsets[i] = off;
      int tiles = (c + BM - 1) / BM;
      for (int m = 0; m < tiles; ++m) { schedule[nt] = make_int2(i, off + m * BM); ++nt; }
      off += tiles * BM;
    }
    *ntiles = nt;
  }
}

__global__ __launch_bounds__(256) void scatter_kernel(
    const int* __restrict__ topk, const int* __restrict__ offsets,
    int* __restrict__ fill, int* __restrict__ row_token, int* __restrict__ row_slot) {
  int tid = blockIdx.x * blockDim.x + threadIdx.x;
  if (tid >= TOKENS * TOPK) return;
  int t = tid >> 2;
  int e = topk[tid];
  int p = atomicAdd(&fill[e], 1);
  int pos = offsets[e] + p;
  row_token[pos] = t;
  row_slot[pos] = tid;
}

// ---------------------------------------------------------------------------
// Fused gate+up GEMM, SiLU epilogue. Weights pre-transposed: Wg/Wu = [N][K].
// Tile 64x128x32, 8 waves, wave = 16x64 strip. Double-buffered async staging.
// schedule==null -> shared-expert mode (identity rows, weight set 0).
// ---------------------------------------------------------------------------
__global__ __launch_bounds__(NTHREADS) void gemm_gu_kernel(
    const bf16* __restrict__ X, const bf16* __restrict__ Wg, const bf16* __restrict__ Wu,
    bf16* __restrict__ out, const int* __restrict__ row_token,
    const int2* __restrict__ schedule, const int* __restrict__ ntiles,
    int nrows, int Kdim, int ncols) {
  __shared__ bf16 As[2][BM * LDSPAD];
  __shared__ bf16 Bgs[2][BN * LDSPAD];
  __shared__ bf16 Bus[2][BN * LDSPAD];

  int e, r0;
  if (schedule) {
    if ((int)blockIdx.x >= *ntiles) return;
    int2 sch = schedule[blockIdx.x];
    e = sch.x; r0 = sch.y;
  } else {
    e = 0; r0 = blockIdx.x * BM;
  }
  const bf16* wg = Wg + (size_t)e * Kdim * ncols;   // [ncols][Kdim]
  const bf16* wu = Wu + (size_t)e * Kdim * ncols;
  int n0 = blockIdx.y * BN;

  int tid = threadIdx.x;
  int lane = tid & 31, wave = tid >> 5;
  int wm = wave & 3, wn = wave >> 2;

  // per-thread A row (gather via row_token for routed experts)
  int ar = tid >> 2;               // 0..63
  int ac = (tid & 3) * 8;          // 0,8,16,24
  int gr_a = r0 + ar;
  int tokr;
  if (row_token) tokr = (gr_a < nrows) ? row_token[gr_a] : -1;
  else           tokr = (gr_a < nrows) ? gr_a : -1;
  const bf16* xrow = (tokr >= 0) ? (X + (size_t)tokr * Kdim) : nullptr;

  auto stage = [&](int buf, int k0) {
    bf16* Asb = As[buf]; bf16* Bgb = Bgs[buf]; bf16* Bub = Bus[buf];
#if HAVE_ASYNC
    if (xrow)
      __builtin_amdgcn_global_load_async_to_lds_b128(AS1P(xrow + k0 + ac),
                                                     AS3P(&Asb[ar * LDSPAD + ac]), 0, 0);
    else
      *(v8bf*)(&Asb[ar * LDSPAD + ac]) = zero8bf();
#else
    v8bf av = zero8bf();
    if (xrow) av = *(const v8bf*)(xrow + k0 + ac);
    *(v8bf*)(&Asb[ar * LDSPAD + ac]) = av;
#endif
#pragma unroll
    for (int l = 0; l < 2; ++l) {
      int c = tid + l * NTHREADS;      // 0..511
      int bn = c >> 2;                 // 0..127
      int kc = (c & 3) * 8;            // 0,8,16,24
      const bf16* sg = wg + (size_t)(n0 + bn) * Kdim + k0 + kc;
      const bf16* su = wu + (size_t)(n0 + bn) * Kdim + k0 + kc;
#if HAVE_ASYNC
      __builtin_amdgcn_global_load_async_to_lds_b128(AS1P(sg),
                                                     AS3P(&Bgb[bn * LDSPAD + kc]), 0, 0);
      __builtin_amdgcn_global_load_async_to_lds_b128(AS1P(su),
                                                     AS3P(&Bub[bn * LDSPAD + kc]), 0, 0);
#else
      *(v8bf*)(&Bgb[bn * LDSPAD + kc]) = *(const v8bf*)sg;
      *(v8bf*)(&Bub[bn * LDSPAD + kc]) = *(const v8bf*)su;
#endif
    }
  };

  v8f accg[4], accu[4];
#pragma unroll
  for (int s = 0; s < 4; ++s) { accg[s] = zero8f(); accu[s] = zero8f(); }

  int kiters = Kdim / BK;
  stage(0, 0);
  wait_async_lds();
  __syncthreads();

  int buf = 0;
  for (int kt = 0; kt < kiters; ++kt) {
    if (kt + 1 < kiters) stage(buf ^ 1, (kt + 1) * BK);  // overlap with WMMA below

    const bf16* Asb = As[buf]; const bf16* Bgb = Bgs[buf]; const bf16* Bub = Bus[buf];
    int arow = wm * 16 + (lane & 15);
    int kb = (lane < 16) ? 0 : 8;
    v16bf afrag = load_frag(&Asb[arow * LDSPAD], kb);
    v16bf bg[4], bu[4];
#pragma unroll
    for (int s = 0; s < 4; ++s) {
      int bn = wn * 64 + s * 16 + (lane & 15);
      bg[s] = load_frag(&Bgb[bn * LDSPAD], kb);
      bu[s] = load_frag(&Bub[bn * LDSPAD], kb);
    }
#pragma unroll
    for (int s = 0; s < 4; ++s) {
      accg[s] = __builtin_amdgcn_wmma_f32_16x16x32_bf16(false, afrag, false, bg[s],
                                                        (short)0, accg[s], false, false);
      accu[s] = __builtin_amdgcn_wmma_f32_16x16x32_bf16(false, afrag, false, bu[s],
                                                        (short)0, accu[s], false, false);
    }
    wait_async_lds();
    __syncthreads();
    buf ^= 1;
  }

  // epilogue: silu(g)*u -> bf16. C layout: lane=(m/8)*16+n, VGPR v -> m=(lane>=16)*8+v
  int crow = wm * 16 + ((lane >> 4) << 3);
  int cn = lane & 15;
#pragma unroll
  for (int s = 0; s < 4; ++s) {
#pragma unroll
    for (int vv = 0; vv < 8; ++vv) {
      float g = accg[s][vv], u = accu[s][vv];
      float a = g * (1.f / (1.f + __expf(-g))) * u;
      int gr = r0 + crow + vv;
      int col = n0 + wn * 64 + s * 16 + cn;
      if (gr < nrows) out[(size_t)gr * ncols + col] = (bf16)a;
    }
  }
}

// ---------------------------------------------------------------------------
// Down-projection GEMM. W pre-transposed [N][K]. Routed mode: weighted atomic
// scatter-add back to token rows; shared mode: plain accumulate into y.
// ---------------------------------------------------------------------------
__global__ __launch_bounds__(NTHREADS) void gemm_down_kernel(
    const bf16* __restrict__ X, const bf16* __restrict__ W, float* __restrict__ y,
    const int* __restrict__ row_slot, const float* __restrict__ tw,
    const int2* __restrict__ schedule, const int* __restrict__ ntiles,
    int nrows, int Kdim, int ncols) {
  __shared__ bf16 As[2][BM * LDSPAD];
  __shared__ bf16 Bs[2][BN * LDSPAD];

  int e, r0;
  if (schedule) {
    if ((int)blockIdx.x >= *ntiles) return;
    int2 sch = schedule[blockIdx.x];
    e = sch.x; r0 = sch.y;
  } else {
    e = 0; r0 = blockIdx.x * BM;
  }
  const bf16* w = W + (size_t)e * Kdim * ncols;     // [ncols][Kdim]
  int n0 = blockIdx.y * BN;

  int tid = threadIdx.x;
  int lane = tid & 31, wave = tid >> 5;
  int wm = wave & 3, wn = wave >> 2;

  int ar = tid >> 2;
  int ac = (tid & 3) * 8;
  int gr_a = r0 + ar;
  const bf16* xrow = (gr_a < nrows) ? (X + (size_t)gr_a * Kdim) : nullptr;

  auto stage = [&](int buf, int k0) {
    bf16* Asb = As[buf]; bf16* Bsb = Bs[buf];
#if HAVE_ASYNC
    if (xrow)
      __builtin_amdgcn_global_load_async_to_lds_b128(AS1P(xrow + k0 + ac),
                                                     AS3P(&Asb[ar * LDSPAD + ac]), 0, 0);
    else
      *(v8bf*)(&Asb[ar * LDSPAD + ac]) = zero8bf();
#else
    v8bf av = zero8bf();
    if (xrow) av = *(const v8bf*)(xrow + k0 + ac);
    *(v8bf*)(&Asb[ar * LDSPAD + ac]) = av;
#endif
#pragma unroll
    for (int l = 0; l < 2; ++l) {
      int c = tid + l * NTHREADS;
      int bn = c >> 2;
      int kc = (c & 3) * 8;
      const bf16* sw = w + (size_t)(n0 + bn) * Kdim + k0 + kc;
#if HAVE_ASYNC
      __builtin_amdgcn_global_load_async_to_lds_b128(AS1P(sw),
                                                     AS3P(&Bsb[bn * LDSPAD + kc]), 0, 0);
#else
      *(v8bf*)(&Bsb[bn * LDSPAD + kc]) = *(const v8bf*)sw;
#endif
    }
  };

  v8f acc[4];
#pragma unroll
  for (int s = 0; s < 4; ++s) acc[s] = zero8f();

  int kiters = Kdim / BK;
  stage(0, 0);
  wait_async_lds();
  __syncthreads();

  int buf = 0;
  for (int kt = 0; kt < kiters; ++kt) {
    if (kt + 1 < kiters) stage(buf ^ 1, (kt + 1) * BK);

    const bf16* Asb = As[buf]; const bf16* Bsb = Bs[buf];
    int arow = wm * 16 + (lane & 15);
    int kb = (lane < 16) ? 0 : 8;
    v16bf afrag = load_frag(&Asb[arow * LDSPAD], kb);
    v16bf bfr[4];
#pragma unroll
    for (int s = 0; s < 4; ++s) {
      int bn = wn * 64 + s * 16 + (lane & 15);
      bfr[s] = load_frag(&Bsb[bn * LDSPAD], kb);
    }
#pragma unroll
    for (int s = 0; s < 4; ++s)
      acc[s] = __builtin_amdgcn_wmma_f32_16x16x32_bf16(false, afrag, false, bfr[s],
                                                       (short)0, acc[s], false, false);
    wait_async_lds();
    __syncthreads();
    buf ^= 1;
  }

  int crow = wm * 16 + ((lane >> 4) << 3);
  int cn = lane & 15;
#pragma unroll
  for (int s = 0; s < 4; ++s) {
#pragma unroll
    for (int vv = 0; vv < 8; ++vv) {
      float val = acc[s][vv];
      int gr = r0 + crow + vv;
      int col = n0 + wn * 64 + s * 16 + cn;
      if (gr >= nrows) continue;
      if (row_slot) {
        int slot = row_slot[gr];
        if (slot >= 0) {
          float wgt = tw[slot];
          atomicAdd(&y[(size_t)(slot >> 2) * ncols + col], wgt * val);
        }
      } else {
        y[(size_t)gr * ncols + col] += val;
      }
    }
  }
}

// ---------------------------------------------------------------------------
// host launch
// ---------------------------------------------------------------------------
extern "C" void kernel_launch(void* const* d_in, const int* in_sizes, int n_in,
                              void* d_out, int out_size, void* d_ws, size_t ws_size,
                              hipStream_t stream) {
  const float* hs      = (const float*)d_in[0];
  const float* gw      = (const float*)d_in[1];
  const float* bias    = (const float*)d_in[2];
  const float* w_gate  = (const float*)d_in[3];
  const float* w_up    = (const float*)d_in[4];
  const float* w_down  = (const float*)d_in[5];
  const float* ws_gate = (const float*)d_in[6];
  const float* ws_up   = (const float*)d_in[7];
  const float* ws_down = (const float*)d_in[8];

  float* y      = (float*)d_out;                          // [T, H]
  float* logits = y + (size_t)TOKENS * HDIM;              // [T, E]
  int*   topk   = (int*)(logits + (size_t)TOKENS * NEXP); // [T, K] int32

  char* ws = (char*)d_ws;
  size_t off = 0;
  auto alloc = [&](size_t bytes) -> char* {
    char* p = ws + off;
    off = (off + bytes + 255) & ~(size_t)255;
    return p;
  };
  int*   counts    = (int*)alloc(NEXP * 4);
  int*   fill      = (int*)alloc(NEXP * 4);
  int*   offsets   = (int*)alloc(NEXP * 4);
  int*   ntiles    = (int*)alloc(4);
  int2*  schedule  = (int2*)alloc(MAX_TILES * sizeof(int2));
  int*   row_token = (int*)alloc(CAP * 4);
  int*   row_slot  = (int*)alloc(CAP * 4);
  float* tw        = (float*)alloc((size_t)TOKENS * TOPK * 4);
  bf16*  xb        = (bf16*)alloc((size_t)TOKENS * HDIM * 2);
  bf16*  wg_b      = (bf16*)alloc((size_t)NEXP * HDIM * IDIM * 2);
  bf16*  wu_b      = (bf16*)alloc((size_t)NEXP * HDIM * IDIM * 2);
  bf16*  wd_b      = (bf16*)alloc((size_t)NEXP * IDIM * HDIM * 2);
  bf16*  wsg_b     = (bf16*)alloc((size_t)HDIM * IDIM * 2);
  bf16*  wsu_b     = (bf16*)alloc((size_t)HDIM * IDIM * 2);
  bf16*  wsd_b     = (bf16*)alloc((size_t)IDIM * HDIM * 2);
  bf16*  h_act     = (bf16*)alloc((size_t)CAP * IDIM * 2);
  bf16*  act_s     = (bf16*)alloc((size_t)TOKENS * IDIM * 2);

  (void)hipMemsetAsync(counts, 0, NEXP * 4, stream);
  (void)hipMemsetAsync(row_token, 0xFF, CAP * 4, stream);
  (void)hipMemsetAsync(row_slot, 0xFF, CAP * 4, stream);
  (void)hipMemsetAsync(y, 0, (size_t)TOKENS * HDIM * 4, stream);

  // activations: straight convert (already K-contiguous)
  {
    int n4 = (int)((size_t)TOKENS * HDIM / 4);
    cvt_bf16_kernel<<<(n4 + 255) / 256, 256, 0, stream>>>(hs, xb, n4);
  }
  // weights: transpose-convert [mat][K][N] -> [mat][N][K]
  auto cvtT = [&](const float* s, bf16* d, int nmat, int K, int N) {
    int total8 = nmat * N * (K / 8);
    cvt_tr_kernel<<<(total8 + 255) / 256, 256, 0, stream>>>(s, d, K, N, total8);
  };
  cvtT(w_gate,  wg_b,  NEXP, HDIM, IDIM);
  cvtT(w_up,    wu_b,  NEXP, HDIM, IDIM);
  cvtT(w_down,  wd_b,  NEXP, IDIM, HDIM);
  cvtT(ws_gate, wsg_b, 1,    HDIM, IDIM);
  cvtT(ws_up,   wsu_b, 1,    HDIM, IDIM);
  cvtT(ws_down, wsd_b, 1,    IDIM, HDIM);

  gate_kernel<<<TOKENS / 8, 256, 0, stream>>>(hs, gw, bias, logits, topk, tw, counts);
  builder_kernel<<<1, 32, 0, stream>>>(counts, offsets, fill, schedule, ntiles);
  scatter_kernel<<<(TOKENS * TOPK + 255) / 256, 256, 0, stream>>>(topk, offsets, fill,
                                                                  row_token, row_slot);

  // routed experts: fused gate+up (SiLU) then weighted down-proj scatter
  gemm_gu_kernel<<<dim3(MAX_TILES, IDIM / BN), NTHREADS, 0, stream>>>(
      xb, wg_b, wu_b, h_act, row_token, schedule, ntiles, CAP, HDIM, IDIM);
  gemm_down_kernel<<<dim3(MAX_TILES, HDIM / BN), NTHREADS, 0, stream>>>(
      h_act, wd_b, y, row_slot, tw, schedule, ntiles, CAP, IDIM, HDIM);

  // shared expert (identity rows, single weight set), accumulates into y
  gemm_gu_kernel<<<dim3(TOKENS / BM, IDIM / BN), NTHREADS, 0, stream>>>(
      xb, wsg_b, wsu_b, act_s, nullptr, nullptr, nullptr, TOKENS, HDIM, IDIM);
  gemm_down_kernel<<<dim3(TOKENS / BM, HDIM / BN), NTHREADS, 0, stream>>>(
      act_s, wsd_b, y, nullptr, nullptr, nullptr, nullptr, TOKENS, IDIM, HDIM);
}
